// ConvAttention_55070070669938
// MI455X (gfx1250) — matrix operand order repaired and checked by reading, not compiled, not run on hardware
//
#include <hip/hip_runtime.h>

// ---------------------------------------------------------------------------
// ConvAttention for MI455X (gfx1250, wave32, WMMA f16 16x16x32 -> f32 accum)
// One 512-thread block (16 waves) per (b,h) pair; grid = 8*64 = 512 blocks.
// Operand orientations chosen so accumulator stores pack into ds_store_b128.
// ---------------------------------------------------------------------------

typedef _Float16 v16h __attribute__((ext_vector_type(16)));
typedef _Float16 v8h  __attribute__((ext_vector_type(8)));
typedef _Float16 v4h  __attribute__((ext_vector_type(4)));
typedef float    v8f  __attribute__((ext_vector_type(8)));

#define LN_EPS      0.001f
#define QK_STRIDE   136      // halves per row for q/k storage (128 + 8 pad)
#define P_STRIDE    264      // halves per row for 256-wide buffers (256 + 8 pad)
#define K_OFF_H     (69632/2)      // half-index offset of K inside region 1
#define R2_OFF      139264        // byte offset of region 2 (W-stage / vT)
#define SMEM_BYTES  274432        // 139264 + 135168  (< 320 KB WGP LDS)

// A-fragment (16x32 f16): per-lane halves 0..7 at k0..k0+7, 8..15 at k0+16..k0+23
__device__ __forceinline__ v16h ld_frag_gap(const _Float16* p) {
  v8h lo = *(const v8h*)(p);
  v8h hi = *(const v8h*)(p + 16);
  v16h r;
#pragma unroll
  for (int i = 0; i < 8; ++i) { r[i] = lo[i]; r[i + 8] = hi[i]; }
  return r;
}

// B-fragment (32x16 f16): per-lane halves 0..7 at k0..k0+7, 8..15 at k0+16..k0+23
__device__ __forceinline__ v16h ld_frag_c(const _Float16* p) {
  v8h lo = *(const v8h*)(p);
  v8h hi = *(const v8h*)(p + 16);
  v16h r;
#pragma unroll
  for (int i = 0; i < 8; ++i) { r[i] = lo[i]; r[i + 8] = hi[i]; }
  return r;
}

#define WMMA_F16(A, B, C) \
  __builtin_amdgcn_wmma_f32_16x16x32_f16(false, (A), false, (B), (short)0, (C), false, false)

__global__ __launch_bounds__(512, 1)
void conv_attn_kernel(const float* __restrict__ x,
                      const float* __restrict__ gamma,
                      const float* __restrict__ beta,
                      const float* __restrict__ Wqkv,   // [256][512] f32
                      float* __restrict__ out)
{
  extern __shared__ char smem[];
  _Float16* R1 = (_Float16*)smem;              // LN-stage / Q+K (stride 136) / P (stride 264)
  _Float16* R2 = (_Float16*)(smem + R2_OFF);   // W-stage / vT (stride 264)

  const int tid  = threadIdx.x;
  const int w    = tid >> 5;          // wave id: owns rows/queries [16w, 16w+16)
  const int lane = tid & 31;
  const int nloc = lane & 15;
  const int hi16 = lane >> 4;         // 0: lanes 0-15, 1: lanes 16-31
  const int bh   = blockIdx.x;
  const size_t rowbase = (size_t)bh * 256;

  // ---- Phase 1: LayerNorm -> xn staged to LDS, packed as B fragments -------
  {
    float4 g0 = ((const float4*)gamma)[lane * 2];
    float4 g1 = ((const float4*)gamma)[lane * 2 + 1];
    float4 b0 = ((const float4*)beta )[lane * 2];
    float4 b1 = ((const float4*)beta )[lane * 2 + 1];
    for (int r = 0; r < 16; ++r) {
      const int m = w * 16 + r;
      const float4* xr = (const float4*)(x + (rowbase + m) * 256);
      float4 a0 = xr[lane * 2];
      float4 a1 = xr[lane * 2 + 1];
      float s = a0.x + a0.y + a0.z + a0.w + a1.x + a1.y + a1.z + a1.w;
      float q = a0.x*a0.x + a0.y*a0.y + a0.z*a0.z + a0.w*a0.w
              + a1.x*a1.x + a1.y*a1.y + a1.z*a1.z + a1.w*a1.w;
#pragma unroll
      for (int mask = 16; mask; mask >>= 1) {
        s += __shfl_xor(s, mask);
        q += __shfl_xor(q, mask);
      }
      const float mu  = s * (1.0f / 256.0f);
      const float var = q * (1.0f / 256.0f) - mu * mu;
      const float rs  = rsqrtf(var + LN_EPS);
      v8h hv;
      hv[0] = (_Float16)((a0.x - mu) * rs * g0.x + b0.x);
      hv[1] = (_Float16)((a0.y - mu) * rs * g0.y + b0.y);
      hv[2] = (_Float16)((a0.z - mu) * rs * g0.z + b0.z);
      hv[3] = (_Float16)((a0.w - mu) * rs * g0.w + b0.w);
      hv[4] = (_Float16)((a1.x - mu) * rs * g1.x + b1.x);
      hv[5] = (_Float16)((a1.y - mu) * rs * g1.y + b1.y);
      hv[6] = (_Float16)((a1.z - mu) * rs * g1.z + b1.z);
      hv[7] = (_Float16)((a1.w - mu) * rs * g1.w + b1.w);
      *(v8h*)(R1 + m * P_STRIDE + lane * 8) = hv;
    }
  }
  // xn as B fragments: lane n <-> row m = 16w+n; halves k0..k0+7 and k0+16..k0+23 per kstep
  v16h xnf[8];
#pragma unroll
  for (int ks = 0; ks < 8; ++ks)
    xnf[ks] = ld_frag_c(R1 + (w * 16 + nloc) * P_STRIDE + ks * 32 + hi16 * 8);

  // ---- W staging: chunk holds f in [256c,256c+256) as f16 [f][c], via 4x4 --
  auto stageW = [&](int chunk) {
    const float* Wc = Wqkv + chunk * 256;
    for (int sb = tid; sb < 4096; sb += 512) {
      const int c0 = (sb >> 6) * 4;
      const int f0 = (sb & 63) * 4;
      float4 r0 = *(const float4*)(Wc + (c0 + 0) * 512 + f0);
      float4 r1 = *(const float4*)(Wc + (c0 + 1) * 512 + f0);
      float4 r2 = *(const float4*)(Wc + (c0 + 2) * 512 + f0);
      float4 r3 = *(const float4*)(Wc + (c0 + 3) * 512 + f0);
      v4h h0, h1, h2, h3;
      h0[0] = (_Float16)r0.x; h0[1] = (_Float16)r1.x; h0[2] = (_Float16)r2.x; h0[3] = (_Float16)r3.x;
      h1[0] = (_Float16)r0.y; h1[1] = (_Float16)r1.y; h1[2] = (_Float16)r2.y; h1[3] = (_Float16)r3.y;
      h2[0] = (_Float16)r0.z; h2[1] = (_Float16)r1.z; h2[2] = (_Float16)r2.z; h2[3] = (_Float16)r3.z;
      h3[0] = (_Float16)r0.w; h3[1] = (_Float16)r1.w; h3[2] = (_Float16)r2.w; h3[3] = (_Float16)r3.w;
      *(v4h*)(R2 + (f0 + 0) * P_STRIDE + c0) = h0;
      *(v4h*)(R2 + (f0 + 1) * P_STRIDE + c0) = h1;
      *(v4h*)(R2 + (f0 + 2) * P_STRIDE + c0) = h2;
      *(v4h*)(R2 + (f0 + 3) * P_STRIDE + c0) = h3;
    }
  };

  stageW(0);
  __syncthreads();

  // ---- Phase 2: q,k = xn @ W[:, :256].  A = W^T tile (rows=f), B = xn ------
  // D rows = f -> packed b128 store of q/k row-major [m][f].
  for (int t = 0; t < 16; ++t) {
    v8f acc = {};
#pragma unroll
    for (int ks = 0; ks < 8; ++ks) {
      v16h a = ld_frag_gap(R2 + (t * 16 + nloc) * P_STRIDE + ks * 32 + hi16 * 8);
      acc = WMMA_F16(a, xnf[ks], acc);
    }
    _Float16* dst = (t < 8) ? R1 : (R1 + K_OFF_H);
    v8h hv;
#pragma unroll
    for (int r = 0; r < 8; ++r) hv[r] = (_Float16)acc[r];
    *(v8h*)(dst + (w * 16 + nloc) * QK_STRIDE + (t & 7) * 16 + hi16 * 8) = hv;
  }
  __syncthreads();

  // ---- Phase 3: v = xn @ W[:, 256:] (accums in regs), store v^T[f][m] ------
  stageW(1);
  __syncthreads();
  v8f vacc[16];
#pragma unroll
  for (int t = 0; t < 16; ++t) {
    v8f acc = {};
#pragma unroll
    for (int ks = 0; ks < 8; ++ks) {
      v16h a = ld_frag_gap(R2 + (t * 16 + nloc) * P_STRIDE + ks * 32 + hi16 * 8);
      acc = WMMA_F16(a, xnf[ks], acc);
    }
    vacc[t] = acc;
  }
  __syncthreads();             // all W-chunk1 reads done -> safe to overwrite
#pragma unroll
  for (int t = 0; t < 16; ++t) {
#pragma unroll
    for (int r = 0; r < 8; ++r) {
      const int f = t * 16 + r + hi16 * 8;
      R2[f * P_STRIDE + w * 16 + nloc] = (_Float16)vacc[t][r];   // vT[f][m]
    }
  }
  __syncthreads();

  // ---- Phase 4: S^T = k q^T  (A = k rows j, B = q strip cols i) ------------
  v8f s[16];
  {
    v16h bq[4];
#pragma unroll
    for (int ks = 0; ks < 4; ++ks)
      bq[ks] = ld_frag_c(R1 + (w * 16 + nloc) * QK_STRIDE + ks * 32 + hi16 * 8);
#pragma unroll
    for (int tj = 0; tj < 16; ++tj) {
      v8f acc = {};
#pragma unroll
      for (int ks = 0; ks < 4; ++ks) {
        v16h a = ld_frag_gap(R1 + K_OFF_H + (tj * 16 + nloc) * QK_STRIDE + ks * 32 + hi16 * 8);
        acc = WMMA_F16(a, bq[ks], acc);
      }
      s[tj] = acc;   // s[tj][r] = score(i = 16w+nloc, j = 16*tj + r + 8*hi16)
    }
  }
  // softmax over j: 128 in-lane values + one partner-lane exchange
  {
    float mx = -3.0e38f;
#pragma unroll
    for (int tj = 0; tj < 16; ++tj)
#pragma unroll
      for (int r = 0; r < 8; ++r) mx = fmaxf(mx, s[tj][r]);
    mx = fmaxf(mx, __shfl_xor(mx, 16));
    float sum = 0.0f;
#pragma unroll
    for (int tj = 0; tj < 16; ++tj)
#pragma unroll
      for (int r = 0; r < 8; ++r) {
        const float e = __expf(s[tj][r] - mx);
        s[tj][r] = e;
        sum += e;
      }
    sum += __shfl_xor(sum, 16);
    const float inv = 1.0f / sum;
#pragma unroll
    for (int tj = 0; tj < 16; ++tj)
#pragma unroll
      for (int r = 0; r < 8; ++r) s[tj][r] *= inv;
  }
  __syncthreads();             // all q/k reads done -> safe to overwrite with P
#pragma unroll
  for (int tj = 0; tj < 16; ++tj) {
    v8h hv;
#pragma unroll
    for (int r = 0; r < 8; ++r) hv[r] = (_Float16)s[tj][r];
    *(v8h*)(R1 + (w * 16 + nloc) * P_STRIDE + tj * 16 + hi16 * 8) = hv;  // P[i][j]
  }

  // ---- Phase 5: y = P @ v, fused residual out = x + y ----------------------
  v16h ap[8];
#pragma unroll
  for (int ks = 0; ks < 8; ++ks)
    ap[ks] = ld_frag_gap(R1 + (w * 16 + nloc) * P_STRIDE + ks * 32 + hi16 * 8);
  for (int tf = 0; tf < 16; ++tf) {
    v8f acc = {};
#pragma unroll
    for (int ks = 0; ks < 8; ++ks) {
      v16h b = ld_frag_c(R2 + (tf * 16 + nloc) * P_STRIDE + ks * 32 + hi16 * 8);
      acc = WMMA_F16(ap[ks], b, acc);
    }
#pragma unroll
    for (int r = 0; r < 8; ++r) {
      const int m = w * 16 + r + hi16 * 8;
      const size_t gi = (rowbase + m) * 256 + tf * 16 + nloc;
      out[gi] = x[gi] + acc[r];
    }
  }
}

extern "C" void kernel_launch(void* const* d_in, const int* in_sizes, int n_in,
                              void* d_out, int out_size, void* d_ws, size_t ws_size,
                              hipStream_t stream) {
  (void)in_sizes; (void)n_in; (void)d_ws; (void)ws_size; (void)out_size;
  const float* x     = (const float*)d_in[0];
  const float* gamma = (const float*)d_in[1];
  const float* beta  = (const float*)d_in[2];
  const float* Wqkv  = (const float*)d_in[3];
  float* out = (float*)d_out;

  hipFuncSetAttribute((const void*)conv_attn_kernel,
                      hipFuncAttributeMaxDynamicSharedMemorySize, SMEM_BYTES);
  conv_attn_kernel<<<512, 512, SMEM_BYTES, stream>>>(x, gamma, beta, Wqkv, out);
}
